// windtest_14276471292809
// MI455X (gfx1250) — matrix operand order repaired and checked by reading, not compiled, not run on hardware
//
#include <hip/hip_runtime.h>
#include <math.h>

// ---------------------------------------------------------------------------
// CDNA5 (gfx1250) wave32 WMMA types
// ---------------------------------------------------------------------------
typedef __attribute__((ext_vector_type(16))) __bf16 v16bf;
typedef __attribute__((ext_vector_type(8)))  float  v8f;

#define DEV static __device__ __forceinline__

DEV unsigned short f2bf(float f) {
  union { float f; unsigned int u; } c; c.f = f;
  unsigned int u = c.u;
  u += 0x7FFFu + ((u >> 16) & 1u);   // round-to-nearest-even
  return (unsigned short)(u >> 16);
}

// B (patch) matrix layout: k-tile-major so each lane's WMMA fragment is
// two aligned 16-byte chunks, uint4 hi and hi+2 (two b128 loads).
// element (k,n) -> ushort[ ((k>>5)*Np + n)*32 + (k&31) ]
DEV void bp_write(unsigned short* Bp, int k, int n, int Np, unsigned short v) {
  Bp[(((size_t)(k >> 5) * Np + n) << 5) + (k & 31)] = v;
}

// ---------------------------------------------------------------------------
// Generic bf16 WMMA GEMM:  Out[M][N] = scale[m]*(A[M][Kp] x B[Kp][N]) + shift[m]
// A: bf16 row-major [M64][Kp] (zero padded). B: k-tile-major (see bp_write).
// 128 threads = 4 waves; wave grid 2x2; wave tile 32x32 via 2x2 WMMA frags.
// Inner loop: 8 global_load_b128 + 4 v_wmma per 32-wide K step.
// ---------------------------------------------------------------------------
__global__ void __launch_bounds__(128)
k_gemm(const unsigned short* __restrict__ A, const unsigned short* __restrict__ Bp,
       void* __restrict__ Out, const float* __restrict__ scale,
       const float* __restrict__ shift,
       int M, int N, int Kp, int Np, int ostride, int relu, int out_bf16) {
  const int lane = threadIdx.x & 31;
  const int wave = threadIdx.x >> 5;
  const int m_base = blockIdx.y * 64 + (wave >> 1) * 32;
  const int n_base = blockIdx.x * 64 + (wave & 1) * 32;
  const int ml = lane & 15;      // M row (A) / N col (B) within 16
  const int hi = lane >> 4;      // half-select per ISA fragment layout
  const int kb = hi * 8;

  const v8f vzero = {0.f,0.f,0.f,0.f,0.f,0.f,0.f,0.f};
  v8f acc[2][2];
#pragma unroll
  for (int i = 0; i < 2; ++i)
#pragma unroll
    for (int j = 0; j < 2; ++j) acc[i][j] = vzero;

  const uint4* A4 = (const uint4*)A;
  const uint4* B4 = (const uint4*)Bp;
  const size_t arow0 = (size_t)(m_base + ml) * Kp;        // halves
  const size_t arow1 = (size_t)(m_base + 16 + ml) * Kp;   // halves
  // B: per (ktile, n) cell = 32 halves = 4 uint4; lane takes uint4 {hi, hi+2}
  size_t bidx0 = (((size_t)(n_base + ml)      ) << 2) + hi;
  size_t bidx1 = (((size_t)(n_base + 16 + ml) ) << 2) + hi;
  const size_t bstep = (size_t)Np << 2;                   // uint4 per k-tile

  for (int k0 = 0; k0 < Kp; k0 += 32) {
    union Frag { v16bf v; uint4 q[2]; } a[2], b[2];
    // A 16x32 bf16 frag: halves [k0+kb, k0+kb+7] and [k0+kb+16, k0+kb+23]
    a[0].q[0] = A4[(arow0 + k0 + kb) >> 3];
    a[0].q[1] = A4[(arow0 + k0 + kb + 16) >> 3];
    a[1].q[0] = A4[(arow1 + k0 + kb) >> 3];
    a[1].q[1] = A4[(arow1 + k0 + kb + 16) >> 3];
    // B 32x16 bf16 frag: two 16-byte chunks per lane (uint4 hi, hi+2) in k-tile-major layout
    b[0].q[0] = B4[bidx0];
    b[0].q[1] = B4[bidx0 + 2];
    b[1].q[0] = B4[bidx1];
    b[1].q[1] = B4[bidx1 + 2];
    bidx0 += bstep;
    bidx1 += bstep;
#pragma unroll
    for (int i = 0; i < 2; ++i)
#pragma unroll
      for (int j = 0; j < 2; ++j)
        acc[i][j] = __builtin_amdgcn_wmma_f32_16x16x32_bf16(
            false, a[i].v, false, b[j].v, (short)0, acc[i][j], false, false);
  }

  float* fo = (float*)Out;
  unsigned short* ho = (unsigned short*)Out;
#pragma unroll
  for (int i = 0; i < 2; ++i) {
#pragma unroll
    for (int j = 0; j < 2; ++j) {
      const int n = n_base + j * 16 + ml;
#pragma unroll
      for (int r = 0; r < 8; ++r) {
        const int m = m_base + i * 16 + hi * 8 + r;  // C/D layout: M = (lane>=16)*8 + vgpr
        if (m < M && n < N) {
          float v = acc[i][j][r] * scale[m] + shift[m];
          if (relu) v = fmaxf(v, 0.f);
          if (out_bf16) ho[(size_t)m * ostride + n] = f2bf(v);
          else          fo[(size_t)m * ostride + n] = v;
        }
      }
    }
  }
}

// ---------------------------------------------------------------------------
// Weight repack (OIHW f32 -> bf16 [M64][Kp], zero padded). k = (c*KH+dy)*KW+dx
// matches OIHW flattening, so it's a flat copy with pad.
// ---------------------------------------------------------------------------
__global__ void k_prep_w(const float* __restrict__ w, unsigned short* __restrict__ A,
                         int M, int K, int M64, int Kp) {
  long i = (long)blockIdx.x * 256 + threadIdx.x;
  if (i >= (long)M64 * Kp) return;
  int m = (int)(i / Kp), k = (int)(i % Kp);
  unsigned short v = 0;
  if (m < M && k < K) v = f2bf(w[(size_t)m * K + k]);
  A[i] = v;
}

// scale/shift prep: mode 0: conv bias only; 1: conv bias + BN folded; 2: BN only
__global__ void k_prep_ss(const float* bias, const float* g, const float* be,
                          const float* mean, const float* var,
                          float* scale, float* shift, int M, int mode) {
  int m = blockIdx.x * 64 + threadIdx.x;
  if (m >= M) return;
  if (mode == 0) { scale[m] = 1.f; shift[m] = bias[m]; }
  else if (mode == 1) {
    float s = g[m] * rsqrtf(var[m] + 1e-5f);
    scale[m] = s; shift[m] = (bias[m] - mean[m]) * s + be[m];
  } else {
    float s = g[m] * rsqrtf(var[m] + 1e-5f);
    scale[m] = s; shift[m] = be[m] - mean[m] * s;
  }
}

// ---------------------------------------------------------------------------
// im2col (image conv): src NCHW (srcCN=0) or CN [c][b*H*W + y*W + x] (srcCN=1)
// ---------------------------------------------------------------------------
__global__ void k_im2col_img(const float* __restrict__ src, unsigned short* __restrict__ Bp,
    int Cin, int H, int W, int OH, int OW, int Bn,
    int KH, int KW, int stride, int dil, int pad,
    int srcCN, int NpSrc, int Kp, int Np, int N) {
  long i = (long)blockIdx.x * 256 + threadIdx.x;
  if (i >= (long)Kp * Np) return;
  int n = (int)(i % Np), k = (int)(i / Np);
  unsigned short v = 0;
  const int K = Cin * KH * KW;
  if (n < N && k < K) {
    const int OHW = OH * OW;
    int b = n / OHW, p = n % OHW;
    int oy = p / OW, ox = p % OW;
    int c = k / (KH * KW), t = k % (KH * KW);
    int y = oy * stride + (t / KW) * dil - pad;
    int x = ox * stride + (t % KW) * dil - pad;
    if (y >= 0 && y < H && x >= 0 && x < W) {
      size_t si = srcCN ? ((size_t)c * NpSrc + (size_t)b * H * W + (size_t)y * W + x)
                        : ((((size_t)b * Cin + c) * H + y) * (size_t)W + x);
      v = f2bf(src[si]);
    }
  }
  bp_write(Bp, k, n, Np, v);
}

// im2col for 3x3 SAME conv inside 7x7 windows, with relative-position-bias fused in.
// N ordering: n = win*49 + ly*7 + lx, win = (b*nW + wy)*nW + wx.
__global__ void k_im2col_win(const float* __restrict__ x0, const float* __restrict__ rpb,
    unsigned short* __restrict__ Bp, int Cin, int Hh, int NpSrc, int Kp, int Np, int N) {
  long i = (long)blockIdx.x * 256 + threadIdx.x;
  if (i >= (long)Kp * Np) return;
  int n = (int)(i % Np), k = (int)(i / Np);
  unsigned short v = 0;
  if (n < N && k < Cin * 9) {
    const int nW = Hh / 7, HW = Hh * Hh;
    int win = n / 49, p = n % 49;
    int ly = p / 7, lx = p % 7;
    int b = win / (nW * nW), wr = win % (nW * nW);
    int wy = wr / nW, wx = wr % nW;
    int c = k / 9, t = k % 9;
    int sy = ly + t / 3 - 1, sx = lx + t % 3 - 1;
    if (sy >= 0 && sy < 7 && sx >= 0 && sx < 7) {   // window-local SAME padding
      int gy = wy * 7 + sy, gx = wx * 7 + sx;
      float val = x0[(size_t)c * NpSrc + (size_t)b * HW + gy * Hh + gx]
                + rpb[(c * 7 + sy) * 7 + sx];
      v = f2bf(val);
    }
  }
  bp_write(Bp, k, n, Np, v);
}

// im2col for the pm conv whose input is relu(ca*ce_w + ce_b) -- computed on the fly.
__global__ void k_im2col_ca(const float* __restrict__ ca, const float* __restrict__ cew,
    const float* __restrict__ ceb, unsigned short* __restrict__ Bp,
    int C2, int Kp, int Np, int N) {
  long i = (long)blockIdx.x * 256 + threadIdx.x;
  if (i >= (long)Kp * Np) return;
  int n = (int)(i % Np), k = (int)(i / Np);
  unsigned short v = 0;
  if (n < N && k < C2 * 9) {
    int win = n / 49, p = n % 49;
    int ly = p / 7, lx = p % 7;
    int c = k / 9, t = k % 9;
    int sy = ly + t / 3 - 1, sx = lx + t % 3 - 1;
    if (sy >= 0 && sy < 7 && sx >= 0 && sx < 7) {
      float val = ca[win * 49 + sy * 7 + sx] * cew[c] + ceb[c];
      v = f2bf(fmaxf(val, 0.f));
    }
  }
  bp_write(Bp, k, n, Np, v);
}

// im2col for gr conv (3x3, dil=2, pad=2) from bf16 gc output [C][NpSrc] (CN layout).
// b0 >= 0: single-image chunk (n = pixel in image b0); b0 < 0: all images.
__global__ void k_im2col_gr(const unsigned short* __restrict__ hgc,
    unsigned short* __restrict__ Bp, int Cin, int Hh, int NpSrc, int b0,
    int Kp, int Np, int N) {
  long i = (long)blockIdx.x * 256 + threadIdx.x;
  if (i >= (long)Kp * Np) return;
  int n = (int)(i % Np), k = (int)(i / Np);
  unsigned short v = 0;
  if (n < N && k < Cin * 9) {
    const int HW = Hh * Hh;
    int b, p;
    if (b0 >= 0) { b = b0; p = n; } else { b = n / HW; p = n % HW; }
    int oy = p / Hh, ox = p % Hh;
    int c = k / 9, t = k % 9;
    int y = oy + (t / 3) * 2 - 2, x = ox + (t % 3) * 2 - 2;
    if (y >= 0 && y < Hh && x >= 0 && x < Hh)
      v = hgc[(size_t)c * NpSrc + (size_t)b * HW + y * Hh + x];
  }
  bp_write(Bp, k, n, Np, v);
}

// out[n] = (sigmoid?) ( sum_m h[m][n]*w[m] + b )
__global__ void k_chan_reduce(const float* __restrict__ h, const float* __restrict__ w,
    const float* __restrict__ bsc, float* __restrict__ out,
    int M, int Np, int N, int do_sig) {
  int n = blockIdx.x * 256 + threadIdx.x;
  if (n >= N) return;
  float s = bsc[0];
  for (int m = 0; m < M; ++m) s += h[(size_t)m * Np + n] * w[m];
  if (do_sig) s = 1.f / (1.f + __expf(-s));
  out[n] = s;
}

// f_att[b*C + c] = sum_p h[c][b*HW+p]*w[p] + b
__global__ void k_feat_reduce(const float* __restrict__ h, const float* __restrict__ w,
    const float* __restrict__ bsc, float* __restrict__ out,
    int Bn, int C, int HW, int Np) {
  int i = blockIdx.x * 64 + threadIdx.x;
  if (i >= Bn * C) return;
  int b = i / C, c = i % C;
  float s = bsc[0];
  for (int p = 0; p < HW; ++p) s += h[(size_t)c * Np + b * HW + p] * w[p];
  out[i] = s;
}

// x1 = BN( pm(window-n order) + 2*x0 + rpb )   (window block double residual)
__global__ void k_win_epi(const float* __restrict__ pm, const float* __restrict__ x0,
    const float* __restrict__ rpb, const float* __restrict__ scale,
    const float* __restrict__ shift, float* __restrict__ x1,
    int C, int Bn, int Hh, int NpG, int NpA) {
  const int HW = Hh * Hh, NT = Bn * HW;
  long i = (long)blockIdx.x * 256 + threadIdx.x;
  if (i >= (long)C * NT) return;
  int c = (int)(i / NT), n = (int)(i % NT);
  int b = n / HW, p = n % HW;
  int y = p / Hh, x = p % Hh;
  const int nW = Hh / 7;
  int win = (b * nW + y / 7) * nW + x / 7;
  int nw = win * 49 + (y % 7) * 7 + (x % 7);
  size_t ai = (size_t)c * NpA + n;
  float v = pm[(size_t)c * NpG + nw] + 2.f * x0[ai] + rpb[(c * 7 + y % 7) * 7 + x % 7];
  x1[ai] = v * scale[c] + shift[c];
}

// x2 = x1 + c_att[b,p] * f_att[b,c]
__global__ void k_glob_epi(const float* __restrict__ x1, const float* __restrict__ catt,
    const float* __restrict__ fatt, float* __restrict__ x2,
    int C, int Bn, int HW, int NpA) {
  const int NT = Bn * HW;
  long i = (long)blockIdx.x * 256 + threadIdx.x;
  if (i >= (long)C * NT) return;
  int c = (int)(i / NT), n = (int)(i % NT);
  int b = n / HW;
  size_t ai = (size_t)c * NpA + n;
  x2[ai] = x1[ai] + catt[n] * fatt[b * C + c];
}

// d_out[b*192 + m] = t[m][b]
__global__ void k_out_tr(const float* __restrict__ t, float* __restrict__ out,
    int Bn, int M, int Np) {
  int i = blockIdx.x * 256 + threadIdx.x;
  if (i >= Bn * M) return;
  int b = i / M, m = i % M;
  out[i] = t[(size_t)m * Np + b];
}

// ---------------------------------------------------------------------------
// Host orchestration
// ---------------------------------------------------------------------------
extern "C" void kernel_launch(void* const* d_in, const int* in_sizes, int n_in,
                              void* d_out, int out_size, void* d_ws, size_t ws_size,
                              hipStream_t stream) {
  (void)in_sizes; (void)n_in; (void)out_size; (void)ws_size;

  const float* IN[64];
  for (int t = 0; t < 61; ++t) IN[t] = (const float*)d_in[t];
  const float* X = IN[0];
  const float *ct_w=IN[1],*ct_b=IN[2],*ct_g=IN[3],*ct_be=IN[4],*ct_m=IN[5],*ct_v=IN[6];
  const float *r1=IN[7],*wcw1=IN[8],*wcb1=IN[9],*caw1=IN[10],*cab1=IN[11],
              *cew1=IN[12],*ceb1=IN[13],*pmw1=IN[14],*pmb1=IN[15],
              *n1g=IN[16],*n1be=IN[17],*n1m=IN[18],*n1v=IN[19],
              *gcw1=IN[20],*gcb1=IN[21],*grw1=IN[22],*grb1=IN[23],
              *gcaw1=IN[24],*gcab1=IN[25],*gfaw1=IN[26],*gfab1=IN[27];
  const float *f1w=IN[28],*f1b=IN[29],*f1g=IN[30],*f1be=IN[31],*f1m=IN[32],*f1v=IN[33];
  const float *r2=IN[34],*wcw2=IN[35],*wcb2=IN[36],*caw2=IN[37],*cab2=IN[38],
              *cew2=IN[39],*ceb2=IN[40],*pmw2=IN[41],*pmb2=IN[42],
              *n2g=IN[43],*n2be=IN[44],*n2m=IN[45],*n2v=IN[46],
              *gcw2=IN[47],*gcb2=IN[48],*grw2=IN[49],*grb2=IN[50],
              *gcaw2=IN[51],*gcab2=IN[52],*gfaw2=IN[53],*gfab2=IN[54];
  const float *f2w=IN[55],*f2b=IN[56],*f2g=IN[57],*f2be=IN[58],*f2m=IN[59],*f2v=IN[60];

  const int NPA = 12544;             // 16*28*28, multiple of 64
  const int NTB = 784;               // 16*7*7
  const int NPB = 832;               // padded to 64
  const float* FN = nullptr;

  // ---- workspace bump allocator ----
  char* wp = (char*)d_ws;
  auto alloc = [&](size_t bytes) -> void* {
    void* r = (void*)wp; wp += (bytes + 255) & ~(size_t)255; return r;
  };
  float* x0  = (float*)alloc((size_t)48 * NPA * 4);
  float* x1  = (float*)alloc((size_t)48 * NPA * 4);
  float* x2  = (float*)alloc((size_t)48 * NPA * 4);
  float* x3  = (float*)alloc((size_t)96 * NPB * 4);
  float* x4  = (float*)alloc((size_t)96 * NPB * 4);
  float* x5  = (float*)alloc((size_t)96 * NPB * 4);
  float* t1  = (float*)alloc((size_t)96 * NPA * 4);            // largest f32 GEMM out
  float* ca  = (float*)alloc((size_t)NPA * 4);
  float* catt= (float*)alloc((size_t)NPA * 4);
  float* fatt= (float*)alloc((size_t)16 * 96 * 4);
  float* tout= (float*)alloc((size_t)192 * 64 * 4);
  unsigned short* hgc = (unsigned short*)alloc((size_t)1536 * NPA * 2);  // gc out, bf16
  unsigned short* Bb  = (unsigned short*)alloc((size_t)12000000 * 2);    // im2col buffer
  unsigned short* A_ct  = (unsigned short*)alloc((size_t)64   * 64    * 2);
  unsigned short* A_wc1 = (unsigned short*)alloc((size_t)128  * 448   * 2);
  unsigned short* A_pm1 = (unsigned short*)alloc((size_t)64   * 864   * 2);
  unsigned short* A_gc1 = (unsigned short*)alloc((size_t)1536 * 448   * 2);
  unsigned short* A_gr1 = (unsigned short*)alloc((size_t)64   * 13824 * 2);
  unsigned short* A_fd1 = (unsigned short*)alloc((size_t)128  * 768   * 2);
  unsigned short* A_wc2 = (unsigned short*)alloc((size_t)192  * 864   * 2);
  unsigned short* A_pm2 = (unsigned short*)alloc((size_t)128  * 1728  * 2);
  unsigned short* A_gc2 = (unsigned short*)alloc((size_t)768  * 864   * 2);
  unsigned short* A_gr2 = (unsigned short*)alloc((size_t)128  * 6912  * 2);
  unsigned short* A_fd2 = (unsigned short*)alloc((size_t)192  * 6144  * 2);
  float* ssp = (float*)alloc((size_t)13 * 4096 * 4);

  auto SS = [&](int i) -> float* { return ssp + (size_t)i * 4096; };
  auto g1 = [](long t) { return dim3((unsigned)((t + 255) / 256)); };

  auto prep = [&](const float* w, const float* bias, const float* g, const float* be,
                  const float* mean, const float* var, unsigned short* A,
                  int ssi, int M, int K, int M64, int Kp, int mode) {
    k_prep_w<<<g1((long)M64 * Kp), 256, 0, stream>>>(w, A, M, K, M64, Kp);
    k_prep_ss<<<dim3((unsigned)((M + 63) / 64)), 64, 0, stream>>>(
        bias, g, be, mean, var, SS(ssi), SS(ssi) + 2048, M, mode);
  };
  auto gemm = [&](const unsigned short* A, void* out, int ssi, int M, int M64,
                  int N, int Kp, int Np, int ostride, int relu, int obf16) {
    dim3 grid(Np / 64, M64 / 64);
    k_gemm<<<grid, 128, 0, stream>>>(A, Bb, out, SS(ssi), SS(ssi) + 2048,
                                     M, N, Kp, Np, ostride, relu, obf16);
  };

  // ---- weight / scale-shift prep (deterministic each call) ----
  prep(ct_w,  ct_b, ct_g, ct_be, ct_m, ct_v, A_ct,  0, 48,   48,   64,   64,   1);
  prep(wcw1,  wcb1, FN,FN,FN,FN,          A_wc1, 1, 96,   432,  128,  448,  0);
  prep(pmw1,  pmb1, FN,FN,FN,FN,          A_pm1, 2, 48,   864,  64,   864,  0);
  prep(gcw1,  gcb1, FN,FN,FN,FN,          A_gc1, 3, 1536, 432,  1536, 448,  0);
  prep(grw1,  grb1, FN,FN,FN,FN,          A_gr1, 4, 48,   13824,64,   13824,0);
  prep(f1w,   f1b,  f1g, f1be, f1m, f1v,  A_fd1, 5, 96,   768,  128,  768,  1);
  prep(wcw2,  wcb2, FN,FN,FN,FN,          A_wc2, 6, 192,  864,  192,  864,  0);
  prep(pmw2,  pmb2, FN,FN,FN,FN,          A_pm2, 7, 96,   1728, 128,  1728, 0);
  prep(gcw2,  gcb2, FN,FN,FN,FN,          A_gc2, 8, 768,  864,  768,  864,  0);
  prep(grw2,  grb2, FN,FN,FN,FN,          A_gr2, 9, 96,   6912, 128,  6912, 0);
  prep(f2w,   f2b,  f2g, f2be, f2m, f2v,  A_fd2,10, 192,  6144, 192,  6144, 1);
  k_prep_ss<<<dim3(1), 64, 0, stream>>>(FN, n1g, n1be, n1m, n1v, SS(11), SS(11)+2048, 48, 2);
  k_prep_ss<<<dim3(2), 64, 0, stream>>>(FN, n2g, n2be, n2m, n2v, SS(12), SS(12)+2048, 96, 2);

  // ---- ChannelTrans: 4x4/s4 conv + BN -> x0 (CN [48][12544]) ----
  k_im2col_img<<<g1((long)64 * NPA), 256, 0, stream>>>(X, Bb, 3, 112, 112, 28, 28, 16,
      4, 4, 4, 1, 0, /*srcCN*/0, 0, 64, NPA, NPA);
  gemm(A_ct, x0, 0, 48, 64, NPA, 64, NPA, NPA, 0, 0);

  // ---- blk1 window attention ----
  k_im2col_win<<<g1((long)448 * NPA), 256, 0, stream>>>(x0, r1, Bb, 48, 28, NPA, 448, NPA, NPA);
  gemm(A_wc1, t1, 1, 96, 128, NPA, 448, NPA, NPA, 1, 0);
  k_chan_reduce<<<g1(NPA), 256, 0, stream>>>(t1, caw1, cab1, ca, 96, NPA, NPA, 1);
  k_im2col_ca<<<g1((long)864 * NPA), 256, 0, stream>>>(ca, cew1, ceb1, Bb, 96, 864, NPA, NPA);
  gemm(A_pm1, t1, 2, 48, 64, NPA, 864, NPA, NPA, 1, 0);
  k_win_epi<<<g1((long)48 * NPA), 256, 0, stream>>>(t1, x0, r1, SS(11), SS(11)+2048, x1,
      48, 16, 28, NPA, NPA);

  // ---- blk1 global attention ----
  k_im2col_img<<<g1((long)448 * NPA), 256, 0, stream>>>(x1, Bb, 48, 28, 28, 28, 28, 16,
      3, 3, 1, 2, 2, /*srcCN*/1, NPA, 448, NPA, NPA);
  gemm(A_gc1, hgc, 3, 1536, 1536, NPA, 448, NPA, NPA, 1, /*bf16*/1);
  for (int b0 = 0; b0 < 16; ++b0) {   // gr conv chunked per image (K=13824)
    k_im2col_gr<<<g1((long)13824 * NPB), 256, 0, stream>>>(hgc, Bb, 1536, 28, NPA, b0,
        13824, NPB, NTB);
    gemm(A_gr1, t1 + (size_t)b0 * NTB, 4, 48, 64, NTB, 13824, NPB, NPA, 1, 0);
  }
  k_chan_reduce<<<g1(NPA), 256, 0, stream>>>(t1, gcaw1, gcab1, catt, 48, NPA, NPA, 0);
  k_feat_reduce<<<dim3((16*48 + 63)/64), 64, 0, stream>>>(t1, gfaw1, gfab1, fatt, 16, 48, 784, NPA);
  k_glob_epi<<<g1((long)48 * NPA), 256, 0, stream>>>(x1, catt, fatt, x2, 48, 16, 784, NPA);

  // ---- FeatureDownload 1: 4x4/s4 conv + BN -> x3 (CN [96][832]) ----
  k_im2col_img<<<g1((long)768 * NPB), 256, 0, stream>>>(x2, Bb, 48, 28, 28, 7, 7, 16,
      4, 4, 4, 1, 0, 1, NPA, 768, NPB, NTB);
  gemm(A_fd1, x3, 5, 96, 128, NTB, 768, NPB, NPB, 0, 0);

  // ---- blk2 window attention ----
  k_im2col_win<<<g1((long)864 * NPB), 256, 0, stream>>>(x3, r2, Bb, 96, 7, NPB, 864, NPB, NTB);
  gemm(A_wc2, t1, 6, 192, 192, NTB, 864, NPB, NPB, 1, 0);
  k_chan_reduce<<<g1(NTB), 256, 0, stream>>>(t1, caw2, cab2, ca, 192, NPB, NTB, 1);
  k_im2col_ca<<<g1((long)1728 * NPB), 256, 0, stream>>>(ca, cew2, ceb2, Bb, 192, 1728, NPB, NTB);
  gemm(A_pm2, t1, 7, 96, 128, NTB, 1728, NPB, NPB, 1, 0);
  k_win_epi<<<g1((long)96 * NTB), 256, 0, stream>>>(t1, x3, r2, SS(12), SS(12)+2048, x4,
      96, 16, 7, NPB, NPB);

  // ---- blk2 global attention ----
  k_im2col_img<<<g1((long)864 * NPB), 256, 0, stream>>>(x4, Bb, 96, 7, 7, 7, 7, 16,
      3, 3, 1, 2, 2, 1, NPB, 864, NPB, NTB);
  gemm(A_gc2, hgc, 8, 768, 768, NTB, 864, NPB, NPB, 1, 1);
  k_im2col_gr<<<g1((long)6912 * NPB), 256, 0, stream>>>(hgc, Bb, 768, 7, NPB, -1, 6912, NPB, NTB);
  gemm(A_gr2, t1, 9, 96, 128, NTB, 6912, NPB, NPB, 1, 0);
  k_chan_reduce<<<g1(NTB), 256, 0, stream>>>(t1, gcaw2, gcab2, catt, 96, NPB, NTB, 0);
  k_feat_reduce<<<dim3((16*96 + 63)/64), 64, 0, stream>>>(t1, gfaw2, gfab2, fatt, 16, 96, 49, NPB);
  k_glob_epi<<<g1((long)96 * NTB), 256, 0, stream>>>(x4, catt, fatt, x5, 96, 16, 49, NPB);

  // ---- FeatureDownload 2: pad 7->8 (via OOB zero), 8x8/s8 conv + BN -> (16,192,1,1) ----
  k_im2col_img<<<g1((long)6144 * 64), 256, 0, stream>>>(x5, Bb, 96, 7, 7, 1, 1, 16,
      8, 8, 8, 1, 0, 1, NPB, 6144, 64, 16);
  gemm(A_fd2, tout, 10, 192, 192, 16, 6144, 64, 64, 0, 0);
  k_out_tr<<<g1(16 * 192), 256, 0, stream>>>(tout, (float*)d_out, 16, 192, 64);
}
